// MaskRcnn_60636348284934
// MI455X (gfx1250) — matrix operand order repaired and checked by reading, not compiled, not run on hardware
//
#include <hip/hip_runtime.h>
#include <math.h>

// ---------------------------------------------------------------------------
// Types for CDNA5 WMMA (wave32): bf16 A/B fragments (16 x bf16 = 8 VGPRs),
// f32 accumulator (8 VGPRs).
// ---------------------------------------------------------------------------
typedef __bf16 bf16_t;
typedef __attribute__((ext_vector_type(8)))  bf16_t v8bf;
typedef __attribute__((ext_vector_type(16))) bf16_t v16bf;
typedef __attribute__((ext_vector_type(8)))  float  v8f;

#define NEGV (-1000000000.0f)

static __device__ __forceinline__ v16bf cat16(v8bf lo, v8bf hi) {
  return __builtin_shufflevector(lo, hi, 0,1,2,3,4,5,6,7,8,9,10,11,12,13,14,15);
}
// A fragment: lane (half,l16) needs K = {half*8 + 0..7} and {half*8 + 16..23}
// -> two aligned 16B LDS loads (row pitch = 40 bf16 = 80B, multiple of 16B).
static __device__ __forceinline__ v16bf ld_frag_a(const bf16_t* row, int half) {
  v8bf lo = *(const v8bf*)(row + half * 8);
  v8bf hi = *(const v8bf*)(row + half * 8 + 16);
  return cat16(lo, hi);
}
// B fragment (stored [n][k] in LDS): lane needs K = {half*8 + 0..7} and {half*8 + 16..23}.
static __device__ __forceinline__ v16bf ld_frag_b(const bf16_t* row, int half) {
  v8bf lo = *(const v8bf*)(row + half * 8);
  v8bf hi = *(const v8bf*)(row + half * 8 + 16);
  return cat16(lo, hi);
}

// ---------------------------------------------------------------------------
// Implicit-GEMM convolution with WMMA bf16 (f32 accumulate).
//   M = OC, K = IC*KH*KW, N = Nb*OH*OW
//   A = weights [OC, IC*KH*KW] (OIHW flat), B = on-the-fly im2col of X (NCHW).
//   Epilogue: v = v*scale[m] + bias[m]; v += residual; relu. Output NCHW.
// Per-thread im2col addressing is hoisted: kt = tid&31 is fixed -> one
// (c,ki,kj) decomposition per K-step; the 16 n-decompositions are computed
// once before the K loop into register tables.
// ---------------------------------------------------------------------------
__global__ __launch_bounds__(256) void conv_wmma(
    const float* __restrict__ X, const float* __restrict__ Wt, float* __restrict__ Out,
    int Nb, int IC, int IH, int IW, int OC, int OH, int OW,
    int KH, int KW, int stride, int pad,
    const float* __restrict__ scale, const float* __restrict__ bias,
    const float* __restrict__ residual, int relu)
{
  constexpr int BM = 128, BN = 128, BK = 32, LDA = 40, LDB = 40;
  __shared__ bf16_t As[BM * LDA];
  __shared__ bf16_t Bs[BN * LDB];

  const int KHW  = KH * KW;
  const int K    = IC * KHW;
  const int OHOW = OH * OW;
  const int N    = Nb * OHOW;
  const int M    = OC;
  const int IHW  = IH * IW;
  const int blockM = blockIdx.y * BM;
  const int blockN = blockIdx.x * BN;
  const int tid  = threadIdx.x;
  const int lane = tid & 31;
  const int half = lane >> 4;
  const int l16  = lane & 15;
  const int wave = tid >> 5;
  const int wRow = (wave >> 1) * 32;   // 4 wave-rows of 32
  const int wCol = (wave & 1) * 64;    // 2 wave-cols of 64

  // ---- hoisted per-thread n-decomposition (fixed for the whole kernel) ----
  const int kt  = tid & 31;  // fixed k-column within tile for this thread
  const int rt0 = tid >> 5;  // first row (both A row mt and B row nt)
  int iy0[16], ix0[16], nbase[16];
#pragma unroll
  for (int i = 0; i < 16; ++i) {
    int gn = blockN + rt0 + i * 8;
    if (gn < N) {
      int b  = gn / OHOW, p = gn % OHOW;
      int oy = p / OW,    ox = p % OW;
      iy0[i]   = oy * stride - pad;
      ix0[i]   = ox * stride - pad;
      nbase[i] = b * IC * IHW;
    } else {
      iy0[i] = 0; ix0[i] = 0; nbase[i] = -1;
    }
  }

  v8f acc[2][4];
  for (int a = 0; a < 2; ++a)
    for (int b = 0; b < 4; ++b)
      for (int i = 0; i < 8; ++i) acc[a][b][i] = 0.0f;

  for (int kk = 0; kk < K; kk += BK) {
    const int gk = kk + kt;
    // one k-decomposition per thread per K-step
    int cbase = 0, ki = 0, kj = 0;
    bool kval = gk < K;
    if (kval) {
      int c = gk / KHW, rem = gk % KHW;
      ki = rem / KW; kj = rem % KW;
      cbase = c * IHW;
    }
    // A tile: weights, row-major [m][k]; prefetch next weight tile column.
    {
      int gmp = blockM + rt0;
      if (kval && kk + BK < K && gmp < M)
        __builtin_prefetch(&Wt[gmp * K + gk + BK], 0, 1);
    }
#pragma unroll
    for (int i = 0; i < 16; ++i) {
      int mt = rt0 + i * 8;
      int gm = blockM + mt;
      float v = (gm < M && kval) ? Wt[gm * K + gk] : 0.0f;
      As[mt * LDA + kt] = (bf16_t)v;
    }
    // B tile: implicit im2col, stored transposed in LDS as [n][k]
#pragma unroll
    for (int i = 0; i < 16; ++i) {
      int nt = rt0 + i * 8;
      float v = 0.0f;
      int iy = iy0[i] + ki;
      int ix = ix0[i] + kj;
      if (kval && nbase[i] >= 0 && iy >= 0 && iy < IH && ix >= 0 && ix < IW)
        v = X[nbase[i] + cbase + iy * IW + ix];
      Bs[nt * LDB + kt] = (bf16_t)v;
    }
    __syncthreads();

    v16bf af[2], bf[4];
#pragma unroll
    for (int tm = 0; tm < 2; ++tm)
      af[tm] = ld_frag_a(As + (wRow + tm * 16 + l16) * LDA, half);
#pragma unroll
    for (int tn = 0; tn < 4; ++tn)
      bf[tn] = ld_frag_b(Bs + (wCol + tn * 16 + l16) * LDB, half);
#pragma unroll
    for (int tm = 0; tm < 2; ++tm)
#pragma unroll
      for (int tn = 0; tn < 4; ++tn)
        acc[tm][tn] = __builtin_amdgcn_wmma_f32_16x16x32_bf16(
            false, af[tm], false, bf[tn], (short)0, acc[tm][tn], false, false);
    __syncthreads();
  }

#pragma unroll
  for (int tm = 0; tm < 2; ++tm) {
#pragma unroll
    for (int tn = 0; tn < 4; ++tn) {
      int n = blockN + wCol + tn * 16 + l16;
      if (n >= N) continue;
      int b = n / OHOW, p = n % OHOW;
#pragma unroll
      for (int r = 0; r < 8; ++r) {
        int m = blockM + wRow + tm * 16 + half * 8 + r;
        if (m >= M) continue;
        float v = acc[tm][tn][r];
        if (scale) v *= scale[m];
        if (bias)  v += bias[m];
        int o = (b * OC + m) * OHOW + p;
        if (residual) v += residual[o];
        if (relu) v = fmaxf(v, 0.0f);
        Out[o] = v;
      }
    }
  }
}

// ---------------------------------------------------------------------------
// Plain GEMM with WMMA bf16: C[M,N] = A[M,K] * B(+bias per n, relu).
// transB: B element = B[n*ldb + k] (for h @ W^T with row-major W[N,K]).
// ---------------------------------------------------------------------------
__global__ __launch_bounds__(256) void gemm_wmma(
    const float* __restrict__ A, const float* __restrict__ B, float* __restrict__ C,
    int M, int N, int K, int lda, int ldb, int ldc,
    int transB, const float* __restrict__ bias_n, int relu)
{
  constexpr int BM = 128, BN = 128, BK = 32, LDA = 40, LDB = 40;
  __shared__ bf16_t As[BM * LDA];
  __shared__ bf16_t Bs[BN * LDB];

  const int blockM = blockIdx.y * BM;
  const int blockN = blockIdx.x * BN;
  const int tid  = threadIdx.x;
  const int lane = tid & 31;
  const int half = lane >> 4;
  const int l16  = lane & 15;
  const int wave = tid >> 5;
  const int wRow = (wave >> 1) * 32;
  const int wCol = (wave & 1) * 64;

  const int kt  = tid & 31;
  const int rt0 = tid >> 5;
  // hoisted per-thread B row bases (n fixed for the whole kernel)
  size_t brow[16];
  bool   bval[16];
#pragma unroll
  for (int i = 0; i < 16; ++i) {
    int gn = blockN + rt0 + i * 8;
    bval[i] = gn < N;
    brow[i] = transB ? (size_t)gn * ldb : (size_t)gn;
  }

  v8f acc[2][4];
  for (int a = 0; a < 2; ++a)
    for (int b = 0; b < 4; ++b)
      for (int i = 0; i < 8; ++i) acc[a][b][i] = 0.0f;

  for (int kk = 0; kk < K; kk += BK) {
    const int gk = kk + kt;
    const bool kval = gk < K;
    {
      int gmp = blockM + rt0;
      if (kval && kk + BK < K && gmp < M)
        __builtin_prefetch(&A[(size_t)gmp * lda + gk + BK], 0, 1);
    }
#pragma unroll
    for (int i = 0; i < 16; ++i) {
      int mt = rt0 + i * 8;
      int gm = blockM + mt;
      float v = (gm < M && kval) ? A[(size_t)gm * lda + gk] : 0.0f;
      As[mt * LDA + kt] = (bf16_t)v;
    }
#pragma unroll
    for (int i = 0; i < 16; ++i) {
      int nt = rt0 + i * 8;
      float v = 0.0f;
      if (bval[i] && kval)
        v = transB ? B[brow[i] + gk] : B[(size_t)gk * ldb + brow[i]];
      Bs[nt * LDB + kt] = (bf16_t)v;
    }
    __syncthreads();

    v16bf af[2], bf[4];
#pragma unroll
    for (int tm = 0; tm < 2; ++tm)
      af[tm] = ld_frag_a(As + (wRow + tm * 16 + l16) * LDA, half);
#pragma unroll
    for (int tn = 0; tn < 4; ++tn)
      bf[tn] = ld_frag_b(Bs + (wCol + tn * 16 + l16) * LDB, half);
#pragma unroll
    for (int tm = 0; tm < 2; ++tm)
#pragma unroll
      for (int tn = 0; tn < 4; ++tn)
        acc[tm][tn] = __builtin_amdgcn_wmma_f32_16x16x32_bf16(
            false, af[tm], false, bf[tn], (short)0, acc[tm][tn], false, false);
    __syncthreads();
  }

#pragma unroll
  for (int tm = 0; tm < 2; ++tm) {
#pragma unroll
    for (int tn = 0; tn < 4; ++tn) {
      int n = blockN + wCol + tn * 16 + l16;
      if (n >= N) continue;
#pragma unroll
      for (int r = 0; r < 8; ++r) {
        int m = blockM + wRow + tm * 16 + half * 8 + r;
        if (m >= M) continue;
        float v = acc[tm][tn][r];
        if (bias_n) v += bias_n[n];
        if (relu) v = fmaxf(v, 0.0f);
        C[(size_t)m * ldc + n] = v;
      }
    }
  }
}

// ---------------------------------------------------------------------------
// Elementwise / glue kernels
// ---------------------------------------------------------------------------
__global__ void fill_k(float* p, float v, int n) {
  int i = blockIdx.x * blockDim.x + threadIdx.x;
  if (i < n) p[i] = v;
}

__global__ void maxpool3x3s2(const float* __restrict__ in, float* __restrict__ out,
                             int C, int IH, int IW, int OH, int OW) {
  int idx = blockIdx.x * blockDim.x + threadIdx.x;
  if (idx >= C * OH * OW) return;
  int ox = idx % OW; int t = idx / OW;
  int oy = t % OH;   int c = t / OH;
  float m = -INFINITY;
  for (int dy = 0; dy < 3; ++dy)
    for (int dx = 0; dx < 3; ++dx) {
      int iy = oy * 2 - 1 + dy, ix = ox * 2 - 1 + dx;
      if (iy >= 0 && iy < IH && ix >= 0 && ix < IW)
        m = fmaxf(m, in[(c * IH + iy) * IW + ix]);
    }
  out[idx] = m;
}

__global__ void upsample_add(float* __restrict__ dst, const float* __restrict__ src,
                             int C, int H, int W) {
  int idx = blockIdx.x * blockDim.x + threadIdx.x;
  if (idx >= C * H * W) return;
  int x = idx % W; int t = idx / W;
  int y = t % H;   int c = t / H;
  dst[idx] += src[(c * (H / 2) + (y >> 1)) * (W / 2) + (x >> 1)];
}

__global__ void subsample2(float* __restrict__ dst, const float* __restrict__ src,
                           int C, int OH, int OW) {
  int idx = blockIdx.x * blockDim.x + threadIdx.x;
  if (idx >= C * OH * OW) return;
  int x = idx % OW; int t = idx / OW;
  int y = t % OH;   int c = t / OH;
  dst[idx] = src[(c * (2 * OH) + 2 * y) * (2 * OW) + 2 * x];
}

__global__ void sigmoid_k(float* x, int n) {
  int i = blockIdx.x * blockDim.x + threadIdx.x;
  if (i < n) x[i] = 1.0f / (1.0f + expf(-x[i]));
}

__global__ void softmax81_k(float* x, int n) {
  int r = blockIdx.x * blockDim.x + threadIdx.x;
  if (r >= n) return;
  float* p = x + r * 81;
  float m = -INFINITY;
  for (int i = 0; i < 81; ++i) m = fmaxf(m, p[i]);
  float s = 0.0f;
  for (int i = 0; i < 81; ++i) { float e = expf(p[i] - m); p[i] = e; s += e; }
  float inv = 1.0f / s;
  for (int i = 0; i < 81; ++i) p[i] *= inv;
}

// RPN per-level: iterative top-k over sigmoid scores [3,H,W] (destroyed),
// decode anchors + deltas for the selected entries.
__global__ __launch_bounds__(1024) void rpn_topk(
    float* __restrict__ cls, const float* __restrict__ reg,
    float* __restrict__ oboxes, float* __restrict__ oscores, float* __restrict__ oids,
    int H, int W, int stride, int K, float lvl)
{
  __shared__ float smax[1024];
  __shared__ int   sidx[1024];
  const int HW = H * W;
  const int L  = 3 * HW;
  const int tid = threadIdx.x;
  const float mr = 4.135166556742356f;  // |log(16/1000)|
  for (int it = 0; it < K; ++it) {
    float best = -INFINITY; int bj = 0;
    for (int i = tid; i < L; i += 1024) {
      int a = i % 3, p = i / 3;
      float s = cls[a * HW + p];
      if (s > best) { best = s; bj = i; }
    }
    smax[tid] = best; sidx[tid] = bj;
    __syncthreads();
    for (int off = 512; off > 0; off >>= 1) {
      if (tid < off && smax[tid + off] > smax[tid]) { smax[tid] = smax[tid + off]; sidx[tid] = sidx[tid + off]; }
      __syncthreads();
    }
    if (tid == 0) {
      int j = sidx[0]; float sj = smax[0];
      int a = j % 3, p = j / 3;
      int y = p / W, x = p % W;
      const float ratios[3] = {0.5f, 1.0f, 2.0f};
      float hr = sqrtf(ratios[a]);
      float ws = stride * (1.0f / hr) * 8.0f;
      float hs = stride * hr * 8.0f;
      float px = (float)(x * stride), py = (float)(y * stride);
      float d0 = reg[(a * 4 + 0) * HW + p];
      float d1 = reg[(a * 4 + 1) * HW + p];
      float dw = fminf(fmaxf(reg[(a * 4 + 2) * HW + p], -mr), mr);
      float dh = fminf(fmaxf(reg[(a * 4 + 3) * HW + p], -mr), mr);
      float gx = px + ws * d0, gy = py + hs * d1;
      float gw = ws * expf(dw), gh = hs * expf(dh);
      oboxes[it * 4 + 0] = fminf(fmaxf(gx - gw * 0.5f, 0.0f), 896.0f);
      oboxes[it * 4 + 1] = fminf(fmaxf(gy - gh * 0.5f, 0.0f), 640.0f);
      oboxes[it * 4 + 2] = fminf(fmaxf(gx + gw * 0.5f, 0.0f), 896.0f);
      oboxes[it * 4 + 3] = fminf(fmaxf(gy + gh * 0.5f, 0.0f), 640.0f);
      oscores[it] = sj;
      oids[it] = lvl;
      cls[a * HW + p] = NEGV;
    }
    __threadfence_block();
    __syncthreads();
  }
}

// Serial-scan NMS (matches reference lax.scan semantics). scores destroyed.
// class_mod != 0 -> group id = i % class_mod ; else gids (or none).
__global__ __launch_bounds__(1024) void nms_k(
    const float* __restrict__ boxes, float* __restrict__ scores,
    const float* __restrict__ gids, int Nb, int K, float thr, int class_mod,
    float* __restrict__ oboxes, float* __restrict__ oscores,
    float* __restrict__ oidx, float* __restrict__ ovalid)
{
  __shared__ float smax[1024];
  __shared__ int   sidx[1024];
  __shared__ float sel[4];
  const int tid = threadIdx.x;
  for (int it = 0; it < K; ++it) {
    float best = -INFINITY; int bj = 0;
    for (int i = tid; i < Nb; i += 1024) {
      float s = scores[i];
      if (s > best) { best = s; bj = i; }
    }
    smax[tid] = best; sidx[tid] = bj;
    __syncthreads();
    for (int off = 512; off > 0; off >>= 1) {
      if (tid < off && smax[tid + off] > smax[tid]) { smax[tid] = smax[tid + off]; sidx[tid] = sidx[tid + off]; }
      __syncthreads();
    }
    int j = sidx[0]; float sj = smax[0];
    if (tid == 0) {
      float g = class_mod ? (float)(j % class_mod) * 2000.0f : (gids ? gids[j] * 2000.0f : 0.0f);
      sel[0] = boxes[j * 4 + 0] + g; sel[1] = boxes[j * 4 + 1] + g;
      sel[2] = boxes[j * 4 + 2] + g; sel[3] = boxes[j * 4 + 3] + g;
      int valid = sj > NEGV * 0.5f;
      oscores[it] = sj;
      oidx[it]    = (float)j;
      ovalid[it]  = valid ? 1.0f : 0.0f;
      oboxes[it * 4 + 0] = valid ? boxes[j * 4 + 0] : 0.0f;
      oboxes[it * 4 + 1] = valid ? boxes[j * 4 + 1] : 0.0f;
      oboxes[it * 4 + 2] = valid ? boxes[j * 4 + 2] : 0.0f;
      oboxes[it * 4 + 3] = valid ? boxes[j * 4 + 3] : 0.0f;
    }
    __syncthreads();
    float bx1 = sel[0], by1 = sel[1], bx2 = sel[2], by2 = sel[3];
    float ba = fmaxf(bx2 - bx1, 0.0f) * fmaxf(by2 - by1, 0.0f);
    for (int i = tid; i < Nb; i += 1024) {
      float g = class_mod ? (float)(i % class_mod) * 2000.0f : (gids ? gids[i] * 2000.0f : 0.0f);
      float x1 = boxes[i * 4 + 0] + g, y1 = boxes[i * 4 + 1] + g;
      float x2 = boxes[i * 4 + 2] + g, y2 = boxes[i * 4 + 3] + g;
      float ix1 = fmaxf(bx1, x1), iy1 = fmaxf(by1, y1);
      float ix2 = fminf(bx2, x2), iy2 = fminf(by2, y2);
      float inter = fmaxf(ix2 - ix1, 0.0f) * fmaxf(iy2 - iy1, 0.0f);
      float ar = fmaxf(x2 - x1, 0.0f) * fmaxf(y2 - y1, 0.0f);
      float iou = inter / (ba + ar - inter + 1e-6f);
      if (iou > thr || i == j) scores[i] = NEGV;
    }
    __threadfence_block();
    __syncthreads();
  }
}

// RoIAlign with per-roi FPN level selection (P2..P5, strides 4..32), sr=2.
__global__ void roi_align_k(
    const float* __restrict__ P2, const float* __restrict__ P3,
    const float* __restrict__ P4, const float* __restrict__ P5,
    const float* __restrict__ rois, int Nroi, int out, float* __restrict__ dst)
{
  int idx = blockIdx.x * blockDim.x + threadIdx.x;
  int total = Nroi * 256 * out * out;
  if (idx >= total) return;
  int ox = idx % out; int t = idx / out;
  int oy = t % out;   t /= out;
  int c  = t % 256;   int r = t / 256;
  float rx1 = rois[r * 4 + 0], ry1 = rois[r * 4 + 1];
  float rx2 = rois[r * 4 + 2], ry2 = rois[r * 4 + 3];
  float wz = rx2 - rx1, hz = ry2 - ry1;
  float sc = sqrtf(fmaxf(wz * hz, 1e-6f));
  float lf = floorf(log2f(sc / 56.0f + 1e-6f));
  int lvl = (int)fminf(fmaxf(lf, 0.0f), 3.0f);
  const float* F; int FH, FW; float fs;
  if (lvl == 0)      { F = P2; FH = 160; FW = 224; fs = 0.25f; }
  else if (lvl == 1) { F = P3; FH = 80;  FW = 112; fs = 0.125f; }
  else if (lvl == 2) { F = P4; FH = 40;  FW = 56;  fs = 0.0625f; }
  else               { F = P5; FH = 20;  FW = 28;  fs = 0.03125f; }
  float x1 = rx1 * fs - 0.5f, y1 = ry1 * fs - 0.5f;
  float x2 = rx2 * fs - 0.5f, y2 = ry2 * fs - 0.5f;
  float bh = (y2 - y1) / out, bw = (x2 - x1) / out;
  const float* feat = F + c * FH * FW;
  float acc = 0.0f;
  for (int s = 0; s < 2; ++s)
    for (int u = 0; u < 2; ++u) {
      float gy = oy + (s + 0.5f) * 0.5f;
      float gx = ox + (u + 0.5f) * 0.5f;
      float ys = y1 + gy * bh, xs = x1 + gx * bw;
      bool valid = (ys > -1.0f) && (ys < (float)FH) && (xs > -1.0f) && (xs < (float)FW);
      float y = fminf(fmaxf(ys, 0.0f), (float)FH - 1.0f);
      float x = fminf(fmaxf(xs, 0.0f), (float)FW - 1.0f);
      int y0 = (int)floorf(y), x0 = (int)floorf(x);
      int y1i = min(y0 + 1, FH - 1), x1i = min(x0 + 1, FW - 1);
      float ly = y - y0, lx = x - x0, hy = 1.0f - ly, hx = 1.0f - lx;
      float v = feat[y0 * FW + x0] * hy * hx + feat[y0 * FW + x1i] * hy * lx +
                feat[y1i * FW + x0] * ly * hx + feat[y1i * FW + x1i] * ly * lx;
      acc += valid ? v : 0.0f;
    }
  dst[idx] = acc * 0.25f;
}

// Second-stage decode: fb[80000,4], fs[80000]
__global__ void final_boxes_k(const float* __restrict__ props, const float* __restrict__ pvalid,
                              const float* __restrict__ probs, const float* __restrict__ regs,
                              float* __restrict__ fb, float* __restrict__ fs)
{
  int idx = blockIdx.x * blockDim.x + threadIdx.x;
  if (idx >= 80000) return;
  int c = idx % 80, r = idx / 80;
  float p0 = props[r * 4 + 0], p1 = props[r * 4 + 1];
  float p2 = props[r * 4 + 2], p3 = props[r * 4 + 3];
  float px = (p0 + p2) * 0.5f, py = (p1 + p3) * 0.5f;
  float pw = p2 - p0, ph = p3 - p1;
  float d0 = regs[r * 320 + c * 4 + 0] * 0.1f;
  float d1 = regs[r * 320 + c * 4 + 1] * 0.1f;
  float d2 = regs[r * 320 + c * 4 + 2] * 0.2f;
  float d3 = regs[r * 320 + c * 4 + 3] * 0.2f;
  const float mr = 4.135166556742356f;
  float dw = fminf(fmaxf(d2, -mr), mr), dh = fminf(fmaxf(d3, -mr), mr);
  float gx = px + pw * d0, gy = py + ph * d1;
  float gw = pw * expf(dw), gh = ph * expf(dh);
  fb[idx * 4 + 0] = fminf(fmaxf(gx - gw * 0.5f, 0.0f), 896.0f);
  fb[idx * 4 + 1] = fminf(fmaxf(gy - gh * 0.5f, 0.0f), 640.0f);
  fb[idx * 4 + 2] = fminf(fmaxf(gx + gw * 0.5f, 0.0f), 896.0f);
  fb[idx * 4 + 3] = fminf(fmaxf(gy + gh * 0.5f, 0.0f), 640.0f);
  float s = probs[r * 81 + c];
  fs[idx] = (pvalid[r] > 0.5f && s > 0.05f) ? s : NEGV;
}

__global__ void write_det_k(const float* __restrict__ db, const float* __restrict__ ds,
                            const float* __restrict__ di, const float* __restrict__ dv,
                            float* __restrict__ out)
{
  int i = threadIdx.x;
  if (i >= 100) return;
  int valid = dv[i] > 0.5f;
  out[i * 5 + 0] = db[i * 4 + 0];
  out[i * 5 + 1] = db[i * 4 + 1];
  out[i * 5 + 2] = db[i * 4 + 2];
  out[i * 5 + 3] = db[i * 4 + 3];
  out[i * 5 + 4] = valid ? ds[i] : 0.0f;
  out[500 + i]   = valid ? (float)(((int)di[i]) % 80) : -1.0f;
}

// deconv2x ('ncij,cohw->noihjw') + bias + relu. x:[100,256,14,14] w:[256,256,2,2]
__global__ void deconv2x_relu_k(const float* __restrict__ x, const float* __restrict__ w,
                                const float* __restrict__ b, float* __restrict__ out)
{
  int idx = blockIdx.x * blockDim.x + threadIdx.x;
  if (idx >= 100 * 256 * 28 * 28) return;
  int ox = idx % 28; int t = idx / 28;
  int oy = t % 28;   t /= 28;
  int o  = t % 256;  int n = t / 256;
  int i = oy >> 1, hh = oy & 1, j = ox >> 1, wi = ox & 1;
  float acc = b[o];
  const float* xp = x + (n * 256) * 196 + i * 14 + j;
  const float* wp = w + o * 4 + hh * 2 + wi;
  for (int c = 0; c < 256; ++c) acc += xp[c * 196] * wp[c * 1024];
  out[idx] = fmaxf(acc, 0.0f);
}

// ---------------------------------------------------------------------------
// Host orchestration
// ---------------------------------------------------------------------------
static inline int cdiv(int a, int b) { return (a + b - 1) / b; }

static void convh(hipStream_t st, const float* X, const float* W, float* O,
                  int Nb, int IC, int IH, int IW, int OC, int OH, int OW,
                  int KH, int KW, int stride, int pad,
                  const float* scale, const float* bias, const float* resid, int relu)
{
  int M = OC, N = Nb * OH * OW;
  dim3 g(cdiv(N, 128), cdiv(M, 128));
  conv_wmma<<<g, 256, 0, st>>>(X, W, O, Nb, IC, IH, IW, OC, OH, OW,
                               KH, KW, stride, pad, scale, bias, resid, relu);
}

static void gemmh(hipStream_t st, const float* A, const float* B, float* C,
                  int M, int N, int K, int lda, int ldb, int ldc,
                  int transB, const float* bias, int relu)
{
  dim3 g(cdiv(N, 128), cdiv(M, 128));
  gemm_wmma<<<g, 256, 0, st>>>(A, B, C, M, N, K, lda, ldb, ldc, transB, bias, relu);
}

struct BlkP {
  const float *bn1b, *bn1s, *bn2b, *bn2s, *bn3b, *bn3s;
  const float *c1, *c2, *c3, *ds, *dsb, *dss;
};

extern "C" void kernel_launch(void* const* d_in, const int* in_sizes, int n_in,
                              void* d_out, int out_size, void* d_ws, size_t ws_size,
                              hipStream_t stream)
{
  (void)in_sizes; (void)n_in; (void)out_size; (void)ws_size;

  // ---- bind inputs in JAX pytree flatten order (dicts sorted by key) ----
  int gi = 0;
  auto nxt = [&]() -> const float* { return (const float*)d_in[gi++]; };

  const float* img = nxt();                       // 'img'
  // params: backbone < bbox < mask < neck < rpn
  // backbone: stages < stem
  const int nblks[4] = {3, 4, 6, 3};
  BlkP blk[16]; int nb = 0;
  for (int s = 0; s < 4; ++s)
    for (int b = 0; b < nblks[s]; ++b) {
      BlkP& B = blk[nb++];
      B.bn1b = nxt(); B.bn1s = nxt();             // bn1: b < s
      B.bn2b = nxt(); B.bn2s = nxt();
      B.bn3b = nxt(); B.bn3s = nxt();
      B.c1 = nxt(); B.c2 = nxt(); B.c3 = nxt();
      if (b == 0) { B.ds = nxt(); B.dsb = nxt(); B.dss = nxt(); }
      else        { B.ds = nullptr; B.dsb = nullptr; B.dss = nullptr; }
    }
  const float* stem_bnb = nxt(); const float* stem_bns = nxt(); const float* stem_w = nxt();
  // bbox: cls_b < cls_w < fc1_b < fc1_w < fc2_b < fc2_w < reg_b < reg_w
  const float* bb_clsb = nxt(); const float* bb_clsw = nxt();
  const float* bb_fc1b = nxt(); const float* bb_fc1w = nxt();
  const float* bb_fc2b = nxt(); const float* bb_fc2w = nxt();
  const float* bb_regb = nxt(); const float* bb_regw = nxt();
  // mask: convs (4 x (w,b)) < logit_b < logit_w < up_b < up_w
  const float *mk_w[4], *mk_b[4];
  for (int i = 0; i < 4; ++i) { mk_w[i] = nxt(); mk_b[i] = nxt(); }
  const float* mk_logitb = nxt(); const float* mk_logitw = nxt();
  const float* mk_upb = nxt();    const float* mk_upw = nxt();
  // neck: fpn (4 x (w,b)) < lateral (4 x (w,b))
  const float *fpn_w[4], *fpn_b[4], *lat_w[4], *lat_b[4];
  for (int i = 0; i < 4; ++i) { fpn_w[i] = nxt(); fpn_b[i] = nxt(); }
  for (int i = 0; i < 4; ++i) { lat_w[i] = nxt(); lat_b[i] = nxt(); }
  // rpn: cls_b < cls_w < conv_b < conv_w < reg_b < reg_w
  const float* rpn_clsb = nxt();  const float* rpn_clsw = nxt();
  const float* rpn_convb = nxt(); const float* rpn_convw = nxt();
  const float* rpn_regb = nxt();  const float* rpn_regw = nxt();

  // ---- workspace bump allocator ----
  char* wp = (char*)d_ws;
  auto alloc = [&](size_t nfloat) -> float* {
    float* p = (float*)wp; wp += nfloat * sizeof(float); return p;
  };
  const int HW1 = 160 * 224, HW2 = 80 * 112, HW3 = 40 * 56, HW4 = 20 * 28, HW5 = 10 * 14;

  float* stem_out = alloc((size_t)64 * 320 * 448);
  float* pooled   = alloc((size_t)64 * HW1);
  float* bufA = alloc((size_t)256 * HW1);
  float* bufB = alloc((size_t)256 * HW1);
  float* tmp1 = alloc((size_t)128 * HW1);
  float* tmp2 = alloc((size_t)128 * HW1);
  float* idn  = alloc((size_t)256 * HW1);
  float* c2s = alloc((size_t)256 * HW1);
  float* c3s = alloc((size_t)512 * HW2);
  float* c4s = alloc((size_t)1024 * HW3);
  float* c5s = alloc((size_t)2048 * HW4);
  float* lat0 = alloc((size_t)256 * HW1);
  float* lat1 = alloc((size_t)256 * HW2);
  float* lat2 = alloc((size_t)256 * HW3);
  float* lat3 = alloc((size_t)256 * HW4);
  float* P2 = alloc((size_t)256 * HW1);
  float* P3 = alloc((size_t)256 * HW2);
  float* P4 = alloc((size_t)256 * HW3);
  float* P5 = alloc((size_t)256 * HW4);
  float* P6 = alloc((size_t)256 * HW5);
  float* clsb  = alloc((size_t)3 * HW1);
  float* regbl = alloc((size_t)12 * HW1);
  float* all_boxes  = alloc(5000 * 4);
  float* all_scores = alloc(5000);
  float* all_ids    = alloc(5000);
  float* props  = alloc(1000 * 4);
  float* pscore = alloc(1000);
  float* pidxb  = alloc(1000);
  float* pvalid = alloc(1000);
  float* roif = alloc((size_t)1000 * 12544);
  float* h1   = alloc((size_t)1000 * 1024);
  float* h2   = alloc((size_t)1000 * 1024);
  float* clsl = alloc((size_t)1000 * 81);
  float* regl = alloc((size_t)1000 * 320);
  float* fb = alloc((size_t)80000 * 4);
  float* fs = alloc(80000);
  float* dboxes  = alloc(100 * 4);
  float* dscores = alloc(100);
  float* didx    = alloc(100);
  float* dvalid  = alloc(100);
  float* mroi = alloc((size_t)100 * 256 * 196);
  float* mA   = alloc((size_t)100 * 256 * 196);
  float* mB   = alloc((size_t)100 * 256 * 196);
  float* mup  = alloc((size_t)100 * 256 * 784);

  // ---- stem + maxpool ----
  convh(stream, img, stem_w, stem_out, 1, 3, 640, 896, 64, 320, 448, 7, 7, 2, 3,
        stem_bns, stem_bnb, nullptr, 1);
  maxpool3x3s2<<<cdiv(64 * HW1, 256), 256, 0, stream>>>(stem_out, pooled, 64, 320, 448, 160, 224);

  // ---- ResNet-50 stages ----
  const float* in = pooled; int inC = 64; int H = 160, W = 224;
  float* saves[4] = {c2s, c3s, c4s, c5s};
  const int planes_a[4] = {64, 128, 256, 512};
  int bgi = 0;
  for (int s = 0; s < 4; ++s) {
    int planes = planes_a[s];
    int stride0 = (s == 0) ? 1 : 2;
    int OH = H / stride0, OW = W / stride0;
    for (int b = 0; b < nblks[s]; ++b) {
      int st = (b == 0) ? stride0 : 1;
      int iH = (b == 0) ? H : OH, iW = (b == 0) ? W : OW;
      float* out = (b == nblks[s] - 1) ? saves[s] : ((b & 1) ? bufB : bufA);
      const BlkP& B = blk[bgi++];
      convh(stream, in, B.c1, tmp1, 1, inC, iH, iW, planes, iH, iW, 1, 1, 1, 0,
            B.bn1s, B.bn1b, nullptr, 1);
      convh(stream, tmp1, B.c2, tmp2, 1, planes, iH, iW, planes, OH, OW, 3, 3, st, 1,
            B.bn2s, B.bn2b, nullptr, 1);
      const float* res;
      if (B.ds) {
        convh(stream, in, B.ds, idn, 1, inC, iH, iW, planes * 4, OH, OW, 1, 1, st, 0,
              B.dss, B.dsb, nullptr, 0);
        res = idn;
      } else res = in;
      convh(stream, tmp2, B.c3, out, 1, planes, OH, OW, planes * 4, OH, OW, 1, 1, 1, 0,
            B.bn3s, B.bn3b, res, 1);
      in = out; inC = planes * 4;
    }
    H = OH; W = OW;
  }

  // ---- FPN ----
  const int Hs[4] = {160, 80, 40, 20}, Ws[4] = {224, 112, 56, 28};
  const int HWs[4] = {HW1, HW2, HW3, HW4};
  const float* csv[4] = {c2s, c3s, c4s, c5s};
  const int cc[4] = {256, 512, 1024, 2048};
  float* latb[4] = {lat0, lat1, lat2, lat3};
  float* Pb[5] = {P2, P3, P4, P5, P6};
  for (int l = 0; l < 4; ++l)
    convh(stream, csv[l], lat_w[l], latb[l], 1, cc[l], Hs[l], Ws[l], 256, Hs[l], Ws[l],
          1, 1, 1, 0, nullptr, lat_b[l], nullptr, 0);
  for (int i = 2; i >= 0; --i)
    upsample_add<<<cdiv(256 * HWs[i], 256), 256, 0, stream>>>(latb[i], latb[i + 1], 256, Hs[i], Ws[i]);
  for (int l = 0; l < 4; ++l)
    convh(stream, latb[l], fpn_w[l], Pb[l], 1, 256, Hs[l], Ws[l], 256, Hs[l], Ws[l],
          3, 3, 1, 1, nullptr, fpn_b[l], nullptr, 0);
  subsample2<<<cdiv(256 * HW5, 256), 256, 0, stream>>>(P6, P5, 256, 10, 14);

  // ---- RPN + per-level top-k ----
  const int lH[5] = {160, 80, 40, 20, 10}, lW[5] = {224, 112, 56, 28, 14};
  const int lS[5] = {4, 8, 16, 32, 64};
  fill_k<<<cdiv(20000, 256), 256, 0, stream>>>(all_boxes, 0.0f, 20000);
  fill_k<<<cdiv(5000, 256), 256, 0, stream>>>(all_scores, NEGV, 5000);
  fill_k<<<cdiv(5000, 256), 256, 0, stream>>>(all_ids, 0.0f, 5000);
  for (int l = 0; l < 5; ++l) {
    int Hh = lH[l], Ww = lW[l], HW = Hh * Ww;
    convh(stream, Pb[l], rpn_convw, bufA, 1, 256, Hh, Ww, 256, Hh, Ww, 3, 3, 1, 1,
          nullptr, rpn_convb, nullptr, 1);
    convh(stream, bufA, rpn_clsw, clsb, 1, 256, Hh, Ww, 3, Hh, Ww, 1, 1, 1, 0,
          nullptr, rpn_clsb, nullptr, 0);
    sigmoid_k<<<cdiv(3 * HW, 256), 256, 0, stream>>>(clsb, 3 * HW);
    convh(stream, bufA, rpn_regw, regbl, 1, 256, Hh, Ww, 12, Hh, Ww, 1, 1, 1, 0,
          nullptr, rpn_regb, nullptr, 0);
    int kl = (3 * HW < 1000) ? 3 * HW : 1000;
    rpn_topk<<<1, 1024, 0, stream>>>(clsb, regbl, all_boxes + l * 4000,
                                     all_scores + l * 1000, all_ids + l * 1000,
                                     Hh, Ww, lS[l], kl, (float)l);
  }

  // ---- proposal NMS ----
  nms_k<<<1, 1024, 0, stream>>>(all_boxes, all_scores, all_ids, 5000, 1000, 0.7f, 0,
                                props, pscore, pidxb, pvalid);

  // ---- RoIAlign (7x7) + bbox head ----
  roi_align_k<<<cdiv(1000 * 256 * 49, 256), 256, 0, stream>>>(P2, P3, P4, P5, props, 1000, 7, roif);
  gemmh(stream, roif, bb_fc1w, h1, 1000, 1024, 12544, 12544, 12544, 1024, 1, bb_fc1b, 1);
  gemmh(stream, h1, bb_fc2w, h2, 1000, 1024, 1024, 1024, 1024, 1024, 1, bb_fc2b, 1);
  gemmh(stream, h2, bb_clsw, clsl, 1000, 81, 1024, 1024, 1024, 81, 1, bb_clsb, 0);
  softmax81_k<<<cdiv(1000, 256), 256, 0, stream>>>(clsl, 1000);
  gemmh(stream, h2, bb_regw, regl, 1000, 320, 1024, 1024, 1024, 320, 1, bb_regb, 0);

  // ---- final decode + class-aware NMS + det outputs ----
  final_boxes_k<<<cdiv(80000, 256), 256, 0, stream>>>(props, pvalid, clsl, regl, fb, fs);
  nms_k<<<1, 1024, 0, stream>>>(fb, fs, nullptr, 80000, 100, 0.5f, 80,
                                dboxes, dscores, didx, dvalid);
  write_det_k<<<1, 128, 0, stream>>>(dboxes, dscores, didx, dvalid, (float*)d_out);

  // ---- mask head ----
  roi_align_k<<<cdiv(100 * 256 * 196, 256), 256, 0, stream>>>(P2, P3, P4, P5, dboxes, 100, 14, mroi);
  const float* min_ = mroi;
  for (int i = 0; i < 4; ++i) {
    float* mout = (i & 1) ? mB : mA;
    convh(stream, min_, mk_w[i], mout, 100, 256, 14, 14, 256, 14, 14, 3, 3, 1, 1,
          nullptr, mk_b[i], nullptr, 1);
    min_ = mout;
  }
  deconv2x_relu_k<<<cdiv(100 * 256 * 784, 256), 256, 0, stream>>>(min_, mk_upw, mk_upb, mup);
  convh(stream, mup, mk_logitw, (float*)d_out + 600, 100, 256, 28, 28, 80, 28, 28,
        1, 1, 1, 0, nullptr, mk_logitb, nullptr, 0);
}
